// PNNModel_12421045420605
// MI455X (gfx1250) — matrix operand order repaired and checked by reading, not compiled, not run on hardware
//
#include <hip/hip_runtime.h>
#include <hip/hip_bf16.h>
#include <stdint.h>
#include <stddef.h>

typedef __attribute__((ext_vector_type(16))) __bf16 v16bf;
typedef __attribute__((ext_vector_type(8)))  __bf16 v8bf;
typedef __attribute__((ext_vector_type(8)))  float  v8f;

#define BATCH  16384
#define NFLD   32
#define EDIM   64
#define VOCAB  100000
#define KREAL1 2544      // 32*64 + 496
#define KPAD1  2560      // padded to multiple of 32
#define N1     1024
#define N2     512
#define NQ     8         // 16x16 n-tiles per wave -> 16(M) x 128(N) wave tile

__device__ __forceinline__ __bf16 f2bf(float f) {
    union { float f; uint32_t u; } v; v.f = f;
    uint32_t r = v.u + 0x7FFFu + ((v.u >> 16) & 1u);   // round to nearest even
    uint16_t h = (uint16_t)(r >> 16);
    return __builtin_bit_cast(__bf16, h);
}
__device__ __forceinline__ float bf2f(__bf16 b) {
    uint32_t u = ((uint32_t)__builtin_bit_cast(uint16_t, b)) << 16;
    return __builtin_bit_cast(float, u);
}

// ---------------------------------------------------------------------------
// Kernel 1: embedding gather + pairwise interaction -> feat (bf16, B x KPAD1)
// One block (256 threads) per batch row.
// ---------------------------------------------------------------------------
__global__ void __launch_bounds__(256)
gather_interact(const int* __restrict__ x, const float* __restrict__ emb,
                __bf16* __restrict__ feat) {
    __shared__ float se[NFLD * EDIM];          // 8 KB
    const int b = blockIdx.x;
    const int t = threadIdx.x;                 // 0..255
    const int f  = t >> 3;                     // field 0..31
    const int d0 = (t & 7) * 8;                // 0,8,..,56

    const int idx = x[b * NFLD + f];
    const float* src = emb + ((size_t)f * VOCAB + (size_t)idx) * EDIM + d0;
    float4 lo = *(const float4*)(src);
    float4 hi = *(const float4*)(src + 4);
    *(float4*)(&se[f * EDIM + d0])     = lo;
    *(float4*)(&se[f * EDIM + d0 + 4]) = hi;

    // embedding portion of feat, as bf16 (16B aligned vector store)
    float vals[8] = {lo.x, lo.y, lo.z, lo.w, hi.x, hi.y, hi.z, hi.w};
    v8bf ev;
    #pragma unroll
    for (int e = 0; e < 8; ++e) ev[e] = f2bf(vals[e]);
    *(v8bf*)(feat + (size_t)b * KPAD1 + f * EDIM + d0) = ev;

    // zero the K padding [2544, 2560)
    if (t < (KPAD1 - KREAL1)) feat[(size_t)b * KPAD1 + KREAL1 + t] = f2bf(0.f);

    __syncthreads();

    const int lane = t & 31;
    for (int p = t; p < 496; p += 256) {
        // map p -> (i, j), numpy triu_indices(32, k=1) order
        int i = 0, rem = p;
        while (rem >= 31 - i) { rem -= 31 - i; ++i; }
        const int j = i + 1 + rem;
        const float* ei = &se[i * EDIM];
        const float* ej = &se[j * EDIM];
        float s = 0.f;
        #pragma unroll
        for (int dd = 0; dd < EDIM; ++dd) {
            int d = (dd + lane) & (EDIM - 1);  // stagger: no LDS bank conflicts
            s += ei[d] * ej[d];
        }
        feat[(size_t)b * KPAD1 + NFLD * EDIM + p] = f2bf(s);
    }
}

// ---------------------------------------------------------------------------
// Kernel 2: pack f32 weight (K x N, row-major) into bf16 WMMA B-fragment
// layout.  Tile = 32(K) x 16(N) = 512 elems, lane L holds col (L&15), elem e
// holds K = (L>>4)*8 + e for e < 8, 16 + (L>>4)*8 + (e-8) for e >= 8 (wave32).
// ---------------------------------------------------------------------------
__global__ void __launch_bounds__(256)
pack_B(const float* __restrict__ W, __bf16* __restrict__ Bp,
       int Kreal, int Kpad, int N) {
    const int t = blockIdx.x * 256 + threadIdx.x;
    if (t >= Kpad * N) return;
    const int tile = t >> 9;
    const int idx  = t & 511;
    const int L = idx >> 4;
    const int e = idx & 15;
    const int ntn = N >> 4;
    const int kt = tile / ntn;
    const int nt = tile % ntn;
    const int n = (nt << 4) + (L & 15);
    const int k = (kt << 5) + ((e < 8) ? ((L >> 4) << 3) + e : 16 + ((L >> 4) << 3) + (e - 8));
    const float v = (k < Kreal) ? W[(size_t)k * N + n] : 0.f;
    Bp[t] = f2bf(v);
}

// ---------------------------------------------------------------------------
// Kernel 3: bf16 GEMM + bias + ReLU via v_wmma_f32_16x16x32_bf16.
// One wave computes a 16(M) x 128(N) tile with 8 accumulators: one A fragment
// feeds 8 WMMAs per K-step (2 A loads + 16 contiguous B loads per 8 WMMAs).
// A: row-major bf16 (M x K, K mult of 32).  Bp: packed B fragments.
// ---------------------------------------------------------------------------
__global__ void __launch_bounds__(256)
gemm_bf16_relu(const __bf16* __restrict__ A, const __bf16* __restrict__ Bp,
               const float* __restrict__ bias, __bf16* __restrict__ Out,
               int M, int N, int K) {
    const int wave = blockIdx.x * (blockDim.x >> 5) + (threadIdx.x >> 5);
    const int lane = threadIdx.x & 31;
    const int ntiles = N >> 7;                        // 128-wide wave tiles
    const int mt = wave / ntiles;
    const int nt = wave % ntiles;
    if (mt >= (M >> 4)) return;
    const int row0 = mt << 4;
    const int col0 = nt << 7;
    const int half = lane >> 4;                       // lane 16..31 -> K+8 group
    const int r = row0 + (lane & 15);

    v8f acc[NQ];
    #pragma unroll
    for (int q = 0; q < NQ; ++q) acc[q] = v8f{};

    const __bf16* Arow = A + (size_t)r * K + half * 8;
    const int ntn = N >> 4;                           // 16-wide n-tiles in packed B

    for (int k0 = 0; k0 < K; k0 += 32) {
        // A fragment: 16-bit A 16x32 layout -> two contiguous b128 loads
        v8bf alo = *(const v8bf*)(Arow + k0);         // K +0..7  (or +8..15)
        v8bf ahi = *(const v8bf*)(Arow + k0 + 16);    // K +16..23 (or +24..31)
        v16bf a = __builtin_shufflevector(alo, ahi,
                    0,1,2,3,4,5,6,7,8,9,10,11,12,13,14,15);

        // 8 B fragments: one contiguous 8 KB block in packed layout
        const size_t tbase =
            ((size_t)(k0 >> 5) * ntn + (col0 >> 4)) * 512 + (size_t)lane * 16;
        v16bf bfrag[NQ];
        #pragma unroll
        for (int q = 0; q < NQ; ++q)
            bfrag[q] = *(const v16bf*)(Bp + tbase + (size_t)q * 512);

        #pragma unroll
        for (int q = 0; q < NQ; ++q)
            acc[q] = __builtin_amdgcn_wmma_f32_16x16x32_bf16(
                false, a, false, bfrag[q], (short)0, acc[q], false, false);
    }

    // C/D layout: VGPR j -> row j (lanes 0-15) / j+8 (lanes 16-31), col = lane&15
    const int nb = col0 + (lane & 15);
    #pragma unroll
    for (int j = 0; j < 8; ++j) {
        const int rr = row0 + j + half * 8;
        __bf16* o = Out + (size_t)rr * N + nb;
        #pragma unroll
        for (int q = 0; q < NQ; ++q) {
            float v = acc[q][j] + bias[nb + q * 16];
            o[q * 16] = f2bf(v > 0.f ? v : 0.f);
        }
    }
}

// ---------------------------------------------------------------------------
// Kernel 4: out[b] = sigmoid(dot(h2[b, :512], W3) + b3), one wave per row.
// ---------------------------------------------------------------------------
__global__ void __launch_bounds__(256)
final_matvec(const __bf16* __restrict__ h2, const float* __restrict__ W3,
             const float* __restrict__ b3, float* __restrict__ out, int M) {
    const int wave = blockIdx.x * (blockDim.x >> 5) + (threadIdx.x >> 5);
    const int lane = threadIdx.x & 31;
    if (wave >= M) return;
    const __bf16* hrow = h2 + (size_t)wave * N2 + lane * 16;
    const float*  wrow = W3 + lane * 16;
    float s = 0.f;
    #pragma unroll
    for (int e = 0; e < 16; ++e) s += bf2f(hrow[e]) * wrow[e];
    #pragma unroll
    for (int off = 16; off > 0; off >>= 1) s += __shfl_xor(s, off, 32);
    if (lane == 0) {
        const float z = s + b3[0];
        out[wave] = 1.f / (1.f + __expf(-z));
    }
}

// ---------------------------------------------------------------------------
extern "C" void kernel_launch(void* const* d_in, const int* in_sizes, int n_in,
                              void* d_out, int out_size, void* d_ws, size_t ws_size,
                              hipStream_t stream) {
    const int*   x   = (const int*)  d_in[0];
    const float* emb = (const float*)d_in[1];
    const float* W1  = (const float*)d_in[2];
    const float* b1  = (const float*)d_in[3];
    const float* W2  = (const float*)d_in[4];
    const float* b2  = (const float*)d_in[5];
    const float* W3  = (const float*)d_in[6];
    const float* b3  = (const float*)d_in[7];
    float* out = (float*)d_out;

    char* ws = (char*)d_ws;
    const size_t feat_bytes = (size_t)BATCH * KPAD1 * 2;   //  83,886,080
    const size_t w1p_bytes  = (size_t)KPAD1 * N1 * 2;      //   5,242,880
    const size_t h1_bytes   = (size_t)BATCH * N1 * 2;      //  33,554,432

    __bf16* feat = (__bf16*)(ws);
    __bf16* W1p  = (__bf16*)(ws + feat_bytes);
    __bf16* h1   = (__bf16*)(ws + feat_bytes + w1p_bytes);
    __bf16* W2p  = (__bf16*)(ws + feat_bytes + w1p_bytes + h1_bytes);
    __bf16* h2   = (__bf16*)(ws);   // aliases feat (dead after GEMM1)

    gather_interact<<<BATCH, 256, 0, stream>>>(x, emb, feat);
    pack_B<<<(KPAD1 * N1 + 255) / 256, 256, 0, stream>>>(W1, W1p, KREAL1, KPAD1, N1);
    pack_B<<<(N1 * N2 + 255) / 256, 256, 0, stream>>>(W2, W2p, N1, N1, N2);

    // GEMM1: 16384x2560 @ 2560x1024 -> waves = 1024 * 8 = 8192 -> 1024 blocks
    gemm_bf16_relu<<<1024, 256, 0, stream>>>(feat, W1p, b1, h1, BATCH, N1, KPAD1);
    // GEMM2: 16384x1024 @ 1024x512 -> waves = 1024 * 4 = 4096 -> 512 blocks
    gemm_bf16_relu<<<512, 256, 0, stream>>>(h1, W2p, b2, h2, BATCH, N2, N1);
    // Final: 16384 rows, 8 waves/block -> 2048 blocks
    final_matvec<<<2048, 256, 0, stream>>>(h2, W3, b3, out, BATCH);
}
